// AttentionModule_53695681135160
// MI455X (gfx1250) — compile-verified, matrix operand order checked by reading, not run on hardware
//
#include <hip/hip_runtime.h>
#include <hip/hip_bf16.h>

// Problem constants
#define BB  2
#define LL  2048
#define DD  1024
#define HH  16
#define HDD 64
#define MMr (BB*LL)   // 4096 token rows

typedef _Float16 h16;
typedef __attribute__((ext_vector_type(16))) _Float16 v16h;
typedef __attribute__((ext_vector_type(8)))  _Float16 v8h;
typedef __attribute__((ext_vector_type(8)))  float    v8f;
typedef __attribute__((ext_vector_type(4)))  float    v4f;

__device__ __forceinline__ v8f zero8() {
  v8f z;
#pragma unroll
  for (int i = 0; i < 8; ++i) z[i] = 0.0f;
  return z;
}

__device__ __forceinline__ v8f wmma_f16(v16h a, v16h b, v8f c) {
  // D = A(16x32 f16) * B(32x16 f16) + C(16x16 f32)
  return __builtin_amdgcn_wmma_f32_16x16x32_f16(false, a, false, b, (short)0, c,
                                                false, false);
}

// Load one 16x32 (or 32x16) f16 fragment. Per the CDNA5 VGPR layout tables:
// lane&15 selects the row (A: M, B: N); lanes >=16 hold the odd K-octet.
// Elements 0..7 = K [kh .. kh+7], elements 8..15 = K [kh+16 .. kh+23],
// kh = (lane>=16) ? 8 : 0.  Two global_load_b128 per fragment per lane.
__device__ __forceinline__ v16h load_frag(const h16* __restrict__ base,
                                          size_t stride, size_t row, int k0,
                                          int lane) {
  int kh = (lane & 16) ? 8 : 0;
  const h16* p = base + row * stride + (size_t)(k0 + kh);
  v8h lo = *(const v8h*)p;
  v8h hi = *(const v8h*)(p + 16);
  v16h f;
#pragma unroll
  for (int i = 0; i < 8; ++i) { f[i] = lo[i]; f[i + 8] = hi[i]; }
  return f;
}

__global__ void cvt_f32_f16(const float* __restrict__ in, h16* __restrict__ out,
                            int n) {
  int i = blockIdx.x * blockDim.x + threadIdx.x;
  int s = gridDim.x * blockDim.x;
  for (; i < n; i += s) out[i] = (h16)in[i];
}

// Y = A(M x K) * W(N x K)^T + bias.  One wave -> 16(M) x 64(N) tile.
// MODE 0: f16 row-major out. MODE 1: f16 transposed out (Vt: (b, d, L)).
// MODE 2: f32 row-major out.
template <int MODE>
__global__ void gemm_wmma(const h16* __restrict__ A, const h16* __restrict__ W,
                          const float* __restrict__ bias, void* __restrict__ out,
                          int Mrows, int N, int K) {
  int lane = threadIdx.x & 31;
  int wave = (int)((blockIdx.x * blockDim.x + threadIdx.x) >> 5);
  int ngrp = N >> 6;                 // N/64 column groups
  int mt = wave / ngrp;
  int ng = wave - mt * ngrp;
  if (mt * 16 >= Mrows) return;      // uniform per wave
  int ln = lane & 15, hi = (lane >> 4) & 1;
  int m0 = mt * 16, n0 = ng * 64;

  v8f acc[4];
#pragma unroll
  for (int g = 0; g < 4; ++g) acc[g] = zero8();

  for (int k0 = 0; k0 < K; k0 += 32) {
    v16h af = load_frag(A, (size_t)K, (size_t)(m0 + ln), k0, lane);
#pragma unroll
    for (int g = 0; g < 4; ++g) {
      v16h bf = load_frag(W, (size_t)K, (size_t)(n0 + g * 16 + ln), k0, lane);
      acc[g] = wmma_f16(af, bf, acc[g]);
    }
  }

#pragma unroll
  for (int g = 0; g < 4; ++g) {
    int col = n0 + g * 16 + ln;
    float bc = bias[col];
    if (MODE == 0) {
      h16* o = (h16*)out;
#pragma unroll
      for (int r = 0; r < 8; ++r)
        o[(size_t)(m0 + 8 * hi + r) * N + col] = (h16)(acc[g][r] + bc);
    } else if (MODE == 1) {
      // transposed store: Vt[(b*DD + col)*LL + l], 8 consecutive l per lane
      h16* o = (h16*)out;
      int b  = m0 >> 11;             // m0 / LL
      int l  = (m0 & (LL - 1)) + 8 * hi;
      v8h pk;
#pragma unroll
      for (int r = 0; r < 8; ++r) pk[r] = (h16)(acc[g][r] + bc);
      *(v8h*)(o + ((size_t)b * DD + col) * LL + l) = pk;
    } else {
      float* o = (float*)out;
#pragma unroll
      for (int r = 0; r < 8; ++r)
        o[(size_t)(m0 + 8 * hi + r) * N + col] = acc[g][r] + bc;
    }
  }
}

// Flash attention, transposed formulation: S^T = K*Q^T, O^T = V^T * P^T.
// One wave handles one (b, h, 16-query tile); j loop in steps of 32 keys.
__global__ void attn_wmma(const h16* __restrict__ Q, const h16* __restrict__ Km,
                          const h16* __restrict__ Vt,
                          const float* __restrict__ bias,
                          const float* __restrict__ mask,
                          h16* __restrict__ outA) {
  const float SCALE = 0.125f;                  // 1/sqrt(64)
  const float LOG2E = 1.4426950408889634f;
  int lane = threadIdx.x & 31;
  int wave = (int)((blockIdx.x * blockDim.x + threadIdx.x) >> 5);
  int qt = wave & 127;                         // LL/16 = 128
  int h  = (wave >> 7) & (HH - 1);
  int b  = wave >> 11;
  int ln = lane & 15, hi = (lane >> 4) & 1;
  int q0 = qt * 16;

  // Q^T B-fragments (lane = q column, elements = hd): natural Q layout.
  size_t rowQ = (size_t)b * LL + q0 + ln;
  v16h qf0 = load_frag(Q, DD, rowQ, h * HDD, lane);
  v16h qf1 = load_frag(Q, DD, rowQ, h * HDD + 32, lane);

  float mrun = -3.0e38f, lrun = 0.0f;
  v8f acc[4];
#pragma unroll
  for (int g = 0; g < 4; ++g) acc[g] = zero8();

  const float* brow = bias + ((((size_t)b * HH + h) * LL) + (q0 + ln)) * LL;
  const float* mrow = mask + ((size_t)b * LL + (q0 + ln)) * LL;
  size_t vrow = (size_t)b * DD + h * HDD;      // base row of V^T slab

  for (int j0 = 0; j0 < LL; j0 += 32) {
    // two 16x16 S^T tiles (keys j0..j0+15, j0+16..j0+31)
    v8f s1 = zero8(), s2 = zero8();
    size_t rowK1 = (size_t)b * LL + j0 + ln;
    v16h ka = load_frag(Km, DD, rowK1, h * HDD, lane);
    v16h kb = load_frag(Km, DD, rowK1, h * HDD + 32, lane);
    s1 = wmma_f16(ka, qf0, s1);
    s1 = wmma_f16(kb, qf1, s1);
    v16h kc = load_frag(Km, DD, rowK1 + 16, h * HDD, lane);
    v16h kd = load_frag(Km, DD, rowK1 + 16, h * HDD + 32, lane);
    s2 = wmma_f16(kc, qf0, s2);
    s2 = wmma_f16(kd, qf1, s2);

    // bias + mask: per lane 8 consecutive j per tile (C-layout rows)
    int jb1 = j0 + 8 * hi;
    int jb2 = j0 + 16 + 8 * hi;
    v4f b1a = *(const v4f*)(brow + jb1), b1b = *(const v4f*)(brow + jb1 + 4);
    v4f b2a = *(const v4f*)(brow + jb2), b2b = *(const v4f*)(brow + jb2 + 4);
    v4f m1a = *(const v4f*)(mrow + jb1), m1b = *(const v4f*)(mrow + jb1 + 4);
    v4f m2a = *(const v4f*)(mrow + jb2), m2b = *(const v4f*)(mrow + jb2 + 4);

    float t[16];
#pragma unroll
    for (int r = 0; r < 4; ++r) {
      t[r]      = (s1[r]     * SCALE + b1a[r] + m1a[r]) * LOG2E;
      t[4 + r]  = (s1[4 + r] * SCALE + b1b[r] + m1b[r]) * LOG2E;
      t[8 + r]  = (s2[r]     * SCALE + b2a[r] + m2a[r]) * LOG2E;
      t[12 + r] = (s2[4 + r] * SCALE + b2b[r] + m2b[r]) * LOG2E;
    }

    // online softmax over this 32-key step (per q column == per lane pair)
    float tmax = t[0];
#pragma unroll
    for (int i = 1; i < 16; ++i) tmax = fmaxf(tmax, t[i]);
    tmax = fmaxf(tmax, __shfl_xor(tmax, 16, 32));
    float mnew = fmaxf(mrun, tmax);
    float corr = exp2f(mrun - mnew);

    v16h pf;                                   // P^T B-fragment, exact slots
    float psum = 0.0f;
#pragma unroll
    for (int r = 0; r < 8; ++r) {
      float p1 = exp2f(t[r] - mnew);
      float p2 = exp2f(t[8 + r] - mnew);
      psum += p1 + p2;
      pf[r] = (h16)p1;
      pf[8 + r] = (h16)p2;
    }
    lrun = lrun * corr + psum;
    mrun = mnew;

#pragma unroll
    for (int g = 0; g < 4; ++g) {
#pragma unroll
      for (int r = 0; r < 8; ++r) acc[g][r] *= corr;
      v16h va = load_frag(Vt, LL, vrow + g * 16 + ln, j0, lane);
      acc[g] = wmma_f16(va, pf, acc[g]);       // O^T += V^T * P^T
    }
  }

  float ltot = lrun + __shfl_xor(lrun, 16, 32);
  float rcp = 1.0f / ltot;
  // O^T tile g: lane = q column, rows = hd -> contiguous d in row-major out
#pragma unroll
  for (int g = 0; g < 4; ++g) {
    v8h pk;
#pragma unroll
    for (int r = 0; r < 8; ++r) pk[r] = (h16)(acc[g][r] * rcp);
    *(v8h*)(outA + ((size_t)b * LL + q0 + ln) * DD + h * HDD + g * 16 + 8 * hi) = pk;
  }
}

extern "C" void kernel_launch(void* const* d_in, const int* in_sizes, int n_in,
                              void* d_out, int out_size, void* d_ws,
                              size_t ws_size, hipStream_t stream) {
  const float* x    = (const float*)d_in[0];
  const float* bias = (const float*)d_in[1];
  const float* mask = (const float*)d_in[2];
  const float* Wq   = (const float*)d_in[3];
  const float* bq   = (const float*)d_in[4];
  const float* Wk   = (const float*)d_in[5];
  const float* bk   = (const float*)d_in[6];
  const float* Wv   = (const float*)d_in[7];
  const float* bv   = (const float*)d_in[8];
  const float* Wo   = (const float*)d_in[9];
  const float* bo   = (const float*)d_in[10];
  float* out = (float*)d_out;

  char* ws = (char*)d_ws;
  const size_t XB = (size_t)MMr * DD * 2;      // 8 MB
  const size_t WB = (size_t)DD * DD * 2;       // 2 MB
  h16* xh   = (h16*)(ws);
  h16* wqh  = (h16*)(ws + XB);
  h16* wkh  = (h16*)(ws + XB + WB);
  h16* wvh  = (h16*)(ws + XB + 2 * WB);
  h16* woh  = (h16*)(ws + XB + 3 * WB);
  h16* Qh   = (h16*)(ws + XB + 4 * WB);
  h16* Kh   = (h16*)(ws + 2 * XB + 4 * WB);
  h16* Vth  = (h16*)(ws + 3 * XB + 4 * WB);
  h16* Ah   = (h16*)(ws + 4 * XB + 4 * WB);    // attention output (f16)

  // fp32 -> fp16 conversions
  cvt_f32_f16<<<1024, 256, 0, stream>>>(x, xh, MMr * DD);
  cvt_f32_f16<<<256, 256, 0, stream>>>(Wq, wqh, DD * DD);
  cvt_f32_f16<<<256, 256, 0, stream>>>(Wk, wkh, DD * DD);
  cvt_f32_f16<<<256, 256, 0, stream>>>(Wv, wvh, DD * DD);
  cvt_f32_f16<<<256, 256, 0, stream>>>(Wo, woh, DD * DD);

  // QKV projections (V stored transposed for the O^T WMMA path)
  const int gemmWaves = (MMr / 16) * (DD / 64);    // 4096
  const int gemmBlocks = gemmWaves / 8;            // 256 thr = 8 waves
  gemm_wmma<0><<<gemmBlocks, 256, 0, stream>>>(xh, wqh, bq, Qh, MMr, DD, DD);
  gemm_wmma<0><<<gemmBlocks, 256, 0, stream>>>(xh, wkh, bk, Kh, MMr, DD, DD);
  gemm_wmma<1><<<gemmBlocks, 256, 0, stream>>>(xh, wvh, bv, Vth, MMr, DD, DD);

  // flash attention: one wave per (b, h, 16-query tile) = 4096 waves
  attn_wmma<<<512, 256, 0, stream>>>(Qh, Kh, Vth, bias, mask, Ah);

  // output projection, f32 result
  gemm_wmma<2><<<gemmBlocks, 256, 0, stream>>>(Ah, woh, bo, out, MMr, DD, DD);
}
